// TransformerDecoderLayerV4_45715631899014
// MI455X (gfx1250) — matrix operand order repaired and checked by reading, not compiled, not run on hardware
//
#include <hip/hip_runtime.h>
#include <hip/hip_bf16.h>
#include <math.h>

// ---------------------------------------------------------------------------
// Problem constants (match reference)
// ---------------------------------------------------------------------------
#define BB   16
#define LQ   256
#define DD   256
#define DFF  1024
#define NH   8
#define DH   32
#define NL   4
#define NP   4
#define SS   21760          // 128^2 + 64^2 + 32^2 + 16^2
#define LKV  272            // NL*NP + LQ

__constant__ int c_lvl_hw[4]  = {128, 64, 32, 16};
__constant__ int c_lvl_off[4] = {0, 16384, 20480, 21504};

typedef __attribute__((ext_vector_type(16))) __bf16 v16bf;
typedef __attribute__((ext_vector_type(8)))  __bf16 v8bf;
typedef __attribute__((ext_vector_type(8)))  float  v8f;

static __device__ __forceinline__ __bf16 f2bf(float f) {
    unsigned u = __float_as_uint(f);
    unsigned r = (u + 0x7FFFu + ((u >> 16) & 1u)) >> 16;   // round-to-nearest-even
    unsigned short s = (unsigned short)r;
    return __builtin_bit_cast(__bf16, s);
}

// Elementwise fp32 -> bf16 (weights are converted once; row-major bf16 is
// already WMMA-B-fragment friendly: a lane's fragment = 16 consecutive K elems)
__global__ void cvt_f32_bf16(const float* __restrict__ in, __bf16* __restrict__ out, int n) {
    int i = blockIdx.x * blockDim.x + threadIdx.x;
    if (i < n) out[i] = f2bf(in[i]);
}

// ---------------------------------------------------------------------------
// WMMA GEMM with async Global->LDS staging of the A strip.
//   out = act(A[M,K] @ W[N,K]^T + bias) (+residual),  W pre-converted to bf16
// Block = 256 threads (8 waves): 16-row strip x nct_block (8/16) column tiles.
// Per 256-float K-chunk: async-copy A fp32 to LDS (ASYNCcnt), one pack pass
// converts it to a bf16 LDS tile, then each wave reuses the LDS A fragment
// for up to 2 column tiles per K-step (2 WMMA issues, no conversion VALU).
// Output row remap: orow = (row / rpb)*obs + (row % rpb) + oro  (row units)
// ---------------------------------------------------------------------------
__global__ void gemm_wmma_bf16(const float* __restrict__ A,
                               const __bf16* __restrict__ Wb,
                               const float* __restrict__ bias,
                               const float* __restrict__ resid,
                               float* __restrict__ out,
                               int M, int N, int K, int relu,
                               int rpb, int obs, int oro, int nct_block) {
    __shared__ float  sA[16 * 256];           // 16 KB fp32 staging (one K-chunk)
    __shared__ __bf16 sAb[16 * 256];          // 8 KB bf16 tile (row-major)
    const int tid  = threadIdx.x;             // 0..255
    const int wid  = tid >> 5;
    const int lane = tid & 31;
    const int row0 = blockIdx.x << 4;
    const int colbase = blockIdx.y * (nct_block << 4);
    const int m  = lane & 15;                 // A-frag M row / B-frag N col
    const int hh = lane >> 4;                 // lane half

    v8f c0 = {};
    v8f c1 = {};
    const float* Abase = A + (size_t)row0 * K;

    for (int kc = 0; kc < K; kc += 256) {
        // ---- async copy: 16 rows x 256 floats (16 KB) -> LDS, 16B per lane op
        #pragma unroll
        for (int i = 0; i < 4; ++i) {
            int chunk = tid + (i << 8);               // 0..1023 (16B chunks)
            int r     = chunk >> 6;                   // 64 chunks per row
            int cko   = (chunk & 63) << 2;            // float offset within row
            const float* gp = Abase + (size_t)r * K + kc + cko;
            unsigned lofs = (unsigned)(uintptr_t)&sA[(r << 8) + cko];
            asm volatile("global_load_async_to_lds_b128 %0, %1, off"
                         :: "v"(lofs), "v"(gp) : "memory");
        }
        asm volatile("s_wait_asynccnt 0x0" ::: "memory");
        __syncthreads();

        // ---- one pack pass: fp32 tile -> bf16 tile (each thread 16 elems)
        {
            int base = tid << 4;
            #pragma unroll
            for (int i = 0; i < 16; ++i) sAb[base + i] = f2bf(sA[base + i]);
        }
        __syncthreads();

        // ---- K-chunk compute: 8 WMMA K-steps of 32
        for (int k0 = 0; k0 < 256; k0 += 32) {
            // A fragment: one contiguous 16-half run from the bf16 LDS tile
            v16bf a = *(const v16bf*)&sAb[(m << 8) + k0 + (hh << 4)];
            // ---- column tile 0: ct = wid (B fragment = 32B contiguous bf16)
            {
                int col0 = colbase + (wid << 4);
                v16bf b = *(const v16bf*)(Wb + (size_t)(col0 + m) * K + kc + k0 + (hh << 4));
                c0 = __builtin_amdgcn_wmma_f32_16x16x32_bf16(false, a, false, b,
                                                             (short)0, c0, false, false);
            }
            // ---- column tile 1: ct = wid + 8 (only when block covers 16 tiles)
            if (nct_block == 16) {
                int col0 = colbase + ((wid + 8) << 4);
                v16bf b = *(const v16bf*)(Wb + (size_t)(col0 + m) * K + kc + k0 + (hh << 4));
                c1 = __builtin_amdgcn_wmma_f32_16x16x32_bf16(false, a, false, b,
                                                             (short)0, c1, false, false);
            }
        }
        __syncthreads();    // protect sA/sAb before next chunk's staging
    }

    // ---- epilogue: C/D layout VGPR r -> M = r + 8*hh; N = lane&15
    const int n = lane & 15;
    #pragma unroll
    for (int t = 0; t < 2; ++t) {
        if (t == 1 && nct_block < 16) break;
        int col0 = colbase + ((wid + (t << 3)) << 4);
        v8f cc = t ? c1 : c0;
        #pragma unroll
        for (int r = 0; r < 8; ++r) {
            int row = row0 + r + (hh << 3);
            int col = col0 + n;
            float v = cc[r];
            if (bias)  v += bias[col];
            size_t orow = (size_t)(row / rpb) * obs + (row % rpb) + oro;
            if (resid) v += resid[orow * N + col];
            if (relu)  v = fmaxf(v, 0.0f);
            out[orow * (size_t)N + col] = v;
        }
    }
}

// ---------------------------------------------------------------------------
// Softmax over the QUERY axis: aw [B, LQ, 128], softmax over LQ per (b, col)
// ---------------------------------------------------------------------------
__global__ void softmax_over_q(float* __restrict__ aw) {
    __shared__ float red[LQ];
    int b = blockIdx.x >> 7;
    int cidx = blockIdx.x & 127;
    int t = threadIdx.x;
    float v = aw[((size_t)(b * LQ + t)) * 128 + cidx];
    red[t] = v; __syncthreads();
    for (int s = 128; s > 0; s >>= 1) { if (t < s) red[t] = fmaxf(red[t], red[t + s]); __syncthreads(); }
    float mx = red[0]; __syncthreads();
    float e = __expf(v - mx);
    red[t] = e; __syncthreads();
    for (int s = 128; s > 0; s >>= 1) { if (t < s) red[t] += red[t + s]; __syncthreads(); }
    aw[((size_t)(b * LQ + t)) * 128 + cidx] = e / red[0];
}

// Softmax over last dim of 16 per (b,q,h): aw [B*LQ*NH, 16]
__global__ void softmax16(float* __restrict__ aw) {
    int idx = blockIdx.x * blockDim.x + threadIdx.x;
    if (idx >= BB * LQ * NH) return;
    float* p = aw + (size_t)idx * 16;
    float mx = -1e30f;
    for (int i = 0; i < 16; ++i) mx = fmaxf(mx, p[i]);
    float s = 0.f, e[16];
    for (int i = 0; i < 16; ++i) { e[i] = __expf(p[i] - mx); s += e[i]; }
    float inv = 1.f / s;
    for (int i = 0; i < 16; ++i) p[i] = e[i] * inv;
}

static __device__ __forceinline__ float bilin(const float* __restrict__ value,
                                              size_t vbase, int H, int W,
                                              float x, float y) {
    float xf = floorf(x), yf = floorf(y);
    int x0 = (int)xf, y0 = (int)yf;
    float wx1 = x - xf, wy1 = y - yf;
    float s = 0.f;
    #pragma unroll
    for (int dy = 0; dy < 2; ++dy) {
        int yy = y0 + dy;
        if (yy < 0 || yy >= H) continue;
        float wy = dy ? wy1 : 1.f - wy1;
        #pragma unroll
        for (int dx = 0; dx < 2; ++dx) {
            int xx = x0 + dx;
            if (xx < 0 || xx >= W) continue;
            float wx = dx ? wx1 : 1.f - wx1;
            s += wx * wy * value[vbase + (size_t)(yy * W + xx) * DD];
        }
    }
    return s;
}

// ---------------------------------------------------------------------------
// Self path sampling: one wave per (b,h,lp); lanes = dh; sum over queries.
// Writes sampled token into rows 0..15 of BOTH kbuf and vbuf [B, LKV, D].
// ---------------------------------------------------------------------------
__global__ void sample_self(const float* __restrict__ value,
                            const float* __restrict__ off1,
                            const float* __restrict__ aw1,
                            float* __restrict__ kbuf, float* __restrict__ vbuf) {
    int wave = (blockIdx.x * blockDim.x + threadIdx.x) >> 5;   // 0..2047
    int lane = threadIdx.x & 31;
    int lp = wave & 15, h = (wave >> 4) & 7, b = wave >> 7;
    int l = lp >> 2;
    int H = c_lvl_hw[l], W = c_lvl_hw[l];
    size_t vbase = ((size_t)b * SS + c_lvl_off[l]) * DD + h * DH + lane;
    float acc = 0.f;
    for (int q = 0; q < LQ; ++q) {
        size_t orow = (size_t)(b * LQ + q) * DD + h * 32 + lp * 2;
        float gx = 2.f * off1[orow]     / (float)W - 1.f;
        float gy = 2.f * off1[orow + 1] / (float)H - 1.f;
        float x = (gx + 1.f) * (W * 0.5f) - 0.5f;
        float y = (gy + 1.f) * (H * 0.5f) - 0.5f;
        float wq = aw1[(size_t)(b * LQ + q) * 128 + h * 16 + lp];
        acc += wq * bilin(value, vbase, H, W, x, y);
    }
    size_t o = ((size_t)b * LKV + lp) * DD + h * DH + lane;
    kbuf[o] = acc;
    vbuf[o] = acc;
}

// ---------------------------------------------------------------------------
// Cross path sampling: one wave per (b,q,h); lanes = dh; sum over (l,p).
// loc = reference_points + off/OFFNORM; grid = 2*loc - 1.
// ---------------------------------------------------------------------------
__global__ void sample_cross(const float* __restrict__ value,
                             const float* __restrict__ off2,
                             const float* __restrict__ aw2,
                             const float* __restrict__ refpts,
                             float* __restrict__ out2) {
    int wave = (blockIdx.x * blockDim.x + threadIdx.x) >> 5;   // 0..32767
    int lane = threadIdx.x & 31;
    int h = wave & 7, q = (wave >> 3) & 255, b = wave >> 11;
    float acc = 0.f;
    for (int lp = 0; lp < 16; ++lp) {
        int l = lp >> 2;
        int H = c_lvl_hw[l], W = c_lvl_hw[l];
        float rx = refpts[((size_t)(b * LQ + q) * NL + l) * 2 + 0];
        float ry = refpts[((size_t)(b * LQ + q) * NL + l) * 2 + 1];
        size_t orow = (size_t)(b * LQ + q) * DD + h * 32 + lp * 2;
        float lx = rx + off2[orow]     / (float)W;
        float ly = ry + off2[orow + 1] / (float)H;
        float gx = 2.f * lx - 1.f, gy = 2.f * ly - 1.f;
        float x = (gx + 1.f) * (W * 0.5f) - 0.5f;
        float y = (gy + 1.f) * (H * 0.5f) - 0.5f;
        float wlp = aw2[(size_t)(b * LQ + q) * 128 + h * 16 + lp];
        size_t vbase = ((size_t)b * SS + c_lvl_off[l]) * DD + h * DH + lane;
        acc += wlp * bilin(value, vbase, H, W, x, y);
    }
    out2[(size_t)(b * LQ + q) * DD + h * DH + lane] = acc;
}

// ---------------------------------------------------------------------------
// Self-attention: one wave per (b,h,q); 272 keys; LDS logits per wave.
// ---------------------------------------------------------------------------
__global__ void attn_272(const float* __restrict__ qh, const float* __restrict__ kh,
                         const float* __restrict__ vh, const float* __restrict__ mask,
                         float* __restrict__ outp) {
    __shared__ float sp[8 * LKV];
    int wid = threadIdx.x >> 5, lane = threadIdx.x & 31;
    int wave = blockIdx.x * 8 + wid;               // 0..32767
    int q = wave & 255, h = (wave >> 8) & 7, b = wave >> 11;
    float* lg = sp + wid * LKV;
    const float scale = 0.17677669529663687f;      // DH^-0.5
    float qv = qh[(size_t)(b * LQ + q) * DD + h * DH + lane];
    for (int j = 0; j < LKV; ++j) {
        float p = qv * kh[((size_t)b * LKV + j) * DD + h * DH + lane];
        p += __shfl_xor(p, 16); p += __shfl_xor(p, 8); p += __shfl_xor(p, 4);
        p += __shfl_xor(p, 2);  p += __shfl_xor(p, 1);
        if (lane == 0) {
            float mv = (j < 16) ? 0.f : mask[(size_t)q * LQ + (j - 16)];
            lg[j] = p * scale + mv;
        }
    }
    __syncthreads();
    float mx = -1e30f;
    for (int j = lane; j < LKV; j += 32) mx = fmaxf(mx, lg[j]);
    mx = fmaxf(mx, __shfl_xor(mx, 16)); mx = fmaxf(mx, __shfl_xor(mx, 8));
    mx = fmaxf(mx, __shfl_xor(mx, 4));  mx = fmaxf(mx, __shfl_xor(mx, 2));
    mx = fmaxf(mx, __shfl_xor(mx, 1));
    float sum = 0.f;
    for (int j = lane; j < LKV; j += 32) { float e = __expf(lg[j] - mx); lg[j] = e; sum += e; }
    sum += __shfl_xor(sum, 16); sum += __shfl_xor(sum, 8); sum += __shfl_xor(sum, 4);
    sum += __shfl_xor(sum, 2);  sum += __shfl_xor(sum, 1);
    float inv = 1.f / sum;
    float o = 0.f;
    for (int j = 0; j < LKV; ++j)
        o += lg[j] * inv * vh[((size_t)b * LKV + j) * DD + h * DH + lane];
    outp[(size_t)(b * LQ + q) * DD + h * DH + lane] = o;
}

// ---------------------------------------------------------------------------
// LayerNorm over D=256: one row per block
// ---------------------------------------------------------------------------
__global__ void layernorm256(const float* __restrict__ x, const float* __restrict__ g,
                             const float* __restrict__ bta, float* __restrict__ y) {
    __shared__ float red[DD];
    int row = blockIdx.x, t = threadIdx.x;
    float v = x[(size_t)row * DD + t];
    red[t] = v; __syncthreads();
    for (int s = 128; s > 0; s >>= 1) { if (t < s) red[t] += red[t + s]; __syncthreads(); }
    float m = red[0] * (1.f / DD); __syncthreads();
    float d = v - m;
    red[t] = d * d; __syncthreads();
    for (int s = 128; s > 0; s >>= 1) { if (t < s) red[t] += red[t + s]; __syncthreads(); }
    float var = red[0] * (1.f / DD);
    y[(size_t)row * DD + t] = d * rsqrtf(var + 1e-5f) * g[t] + bta[t];
}

__global__ void addvec(const float* __restrict__ a, const float* __restrict__ b,
                       float* __restrict__ o, int n) {
    int i = blockIdx.x * blockDim.x + threadIdx.x;
    if (i < n) o[i] = a[i] + b[i];
}

// ---------------------------------------------------------------------------
// Host launch
// ---------------------------------------------------------------------------
extern "C" void kernel_launch(void* const* d_in, const int* in_sizes, int n_in,
                              void* d_out, int out_size, void* d_ws, size_t ws_size,
                              hipStream_t stream) {
    const float* tgt       = (const float*)d_in[0];
    const float* query_pos = (const float*)d_in[1];
    const float* refpts    = (const float*)d_in[2];
    const float* src       = (const float*)d_in[3];
    const float* mask      = (const float*)d_in[4];
    const float* wq        = (const float*)d_in[5];
    const float* wk        = (const float*)d_in[6];
    const float* wv        = (const float*)d_in[7];
    const float* so_w      = (const float*)d_in[8];
    const float* so_b      = (const float*)d_in[9];
    const float* sp_w      = (const float*)d_in[10];
    const float* sp_b      = (const float*)d_in[11];
    const float* aw_w      = (const float*)d_in[12];
    const float* aw_b      = (const float*)d_in[13];
    const float* sa_in_w   = (const float*)d_in[14];
    const float* sa_in_b   = (const float*)d_in[15];
    const float* sa_out_w  = (const float*)d_in[16];
    const float* sa_out_b  = (const float*)d_in[17];
    const float* ca_so_w   = (const float*)d_in[18];
    const float* ca_so_b   = (const float*)d_in[19];
    const float* ca_aw_w   = (const float*)d_in[20];
    const float* ca_aw_b   = (const float*)d_in[21];
    const float* ca_v_w    = (const float*)d_in[22];
    const float* ca_v_b    = (const float*)d_in[23];
    const float* ca_out_w  = (const float*)d_in[24];
    const float* ca_out_b  = (const float*)d_in[25];
    const float* l1_w      = (const float*)d_in[26];
    const float* l1_b      = (const float*)d_in[27];
    const float* l2_w      = (const float*)d_in[28];
    const float* l2_b      = (const float*)d_in[29];
    const float* n1_g = (const float*)d_in[30];
    const float* n1_b = (const float*)d_in[31];
    const float* n2_g = (const float*)d_in[32];
    const float* n2_b = (const float*)d_in[33];
    const float* n3_g = (const float*)d_in[34];
    const float* n3_b = (const float*)d_in[35];

    const int MQ = BB * LQ;        // 4096
    const int MKV = BB * LKV;      // 4352
    const int MV = BB * SS;        // 348160

    float* ws = (float*)d_ws;
    size_t off = 0;
    auto alloc = [&](size_t n) { float* p = ws + off; off += n; return p; };
    float* value = alloc((size_t)MV * DD);      // 89.1M floats (reused by both paths)
    float* qbuf  = alloc((size_t)MQ * DD);      // q (later reused as t2)
    float* kbuf  = alloc((size_t)MKV * DD);
    float* vbuf  = alloc((size_t)MKV * DD);
    float* offb  = alloc((size_t)MQ * DD);      // sampling offsets (both paths)
    float* awb   = alloc((size_t)MQ * 128);     // attention weights (both paths)
    float* qhb   = alloc((size_t)MQ * DD);
    float* khb   = alloc((size_t)MKV * DD);
    float* vhb   = alloc((size_t)MKV * DD);
    float* tmpA  = alloc((size_t)MQ * DD);      // attn out / deform out
    float* t1    = alloc((size_t)MQ * DD);
    float* qc    = alloc((size_t)MQ * DD);
    float* tmpB  = alloc((size_t)MQ * DD);      // pre-LN buffers
    float* ffn   = alloc((size_t)MQ * DFF);

    // bf16 weight pool (converted once per launch)
    __bf16* wsb = (__bf16*)(ws + off);
    size_t boff = 0;
    auto balloc = [&](size_t n) { __bf16* p = wsb + boff; boff += n; return p; };
    __bf16* b_wq     = balloc(DD * DD);
    __bf16* b_wk     = balloc(DD * DD);
    __bf16* b_wv     = balloc(DD * DD);
    __bf16* b_so     = balloc(DD * DD);
    __bf16* b_aw     = balloc(128 * DD);
    __bf16* b_sp     = balloc(DD * DD);
    __bf16* b_sa_in  = balloc(3 * DD * DD);
    __bf16* b_sa_out = balloc(DD * DD);
    __bf16* b_ca_so  = balloc(DD * DD);
    __bf16* b_ca_aw  = balloc(128 * DD);
    __bf16* b_ca_v   = balloc(DD * DD);
    __bf16* b_ca_out = balloc(DD * DD);
    __bf16* b_l1     = balloc(DFF * DD);
    __bf16* b_l2     = balloc(DD * DFF);

    auto cvt = [&](const float* in, __bf16* o, int n) {
        cvt_f32_bf16<<<(n + 255) / 256, 256, 0, stream>>>(in, o, n);
    };
    cvt(wq, b_wq, DD * DD);           cvt(wk, b_wk, DD * DD);
    cvt(wv, b_wv, DD * DD);           cvt(so_w, b_so, DD * DD);
    cvt(aw_w, b_aw, 128 * DD);        cvt(sp_w, b_sp, DD * DD);
    cvt(sa_in_w, b_sa_in, 3 * DD * DD); cvt(sa_out_w, b_sa_out, DD * DD);
    cvt(ca_so_w, b_ca_so, DD * DD);   cvt(ca_aw_w, b_ca_aw, 128 * DD);
    cvt(ca_v_w, b_ca_v, DD * DD);     cvt(ca_out_w, b_ca_out, DD * DD);
    cvt(l1_w, b_l1, DFF * DD);        cvt(l2_w, b_l2, DD * DFF);

    auto gemm = [&](const float* A, const __bf16* W, const float* bias, const float* resid,
                    float* o, int M, int N, int K, int relu, int rpb, int obs, int oro) {
        int nct = N >> 4;                         // column tiles total
        int nct_block = (nct >= 16) ? 16 : nct;   // tiles per block (8 or 16)
        dim3 grid(M / 16, nct / nct_block);
        gemm_wmma_bf16<<<grid, 256, 0, stream>>>(A, W, bias, resid, o,
                                                 M, N, K, relu, rpb, obs, oro, nct_block);
    };

    // ---- projections from tgt ----
    gemm(tgt, b_wq, nullptr, query_pos, qbuf, MQ, DD, DD, 0, MQ, 0, 0);     // q = tgt@wq^T + query_pos
    gemm(tgt, b_wk, nullptr, nullptr, kbuf, MQ, DD, DD, 0, LQ, LKV, 16);    // k -> rows 16.. of kbuf
    gemm(tgt, b_wv, nullptr, nullptr, vbuf, MQ, DD, DD, 0, LQ, LKV, 16);    // v -> rows 16.. of vbuf
    gemm(tgt, b_so, so_b, nullptr, offb, MQ, DD, DD, 0, MQ, 0, 0);          // sampling offsets
    gemm(tgt, b_aw, aw_b, nullptr, awb, MQ, 128, DD, 0, MQ, 0, 0);          // attn-weight logits
    softmax_over_q<<<BB * 128, LQ, 0, stream>>>(awb);

    // ---- value projection (memory-bound big GEMM) + self-path sampling ----
    gemm(src, b_sp, sp_b, nullptr, value, MV, DD, DD, 0, MV, 0, 0);
    sample_self<<<256, 256, 0, stream>>>(value, offb, awb, kbuf, vbuf);     // rows 0..15

    // ---- self attention ----
    gemm(qbuf, b_sa_in,           sa_in_b,       nullptr, qhb, MQ,  DD, DD, 0, MQ,  0, 0);
    gemm(kbuf, b_sa_in + 256 * DD, sa_in_b + 256, nullptr, khb, MKV, DD, DD, 0, MKV, 0, 0);
    gemm(vbuf, b_sa_in + 512 * DD, sa_in_b + 512, nullptr, vhb, MKV, DD, DD, 0, MKV, 0, 0);
    attn_272<<<MQ * NH / 8, 256, 0, stream>>>(qhb, khb, vhb, mask, tmpA);
    gemm(tmpA, b_sa_out, sa_out_b, tgt, tmpB, MQ, DD, DD, 0, MQ, 0, 0);     // + residual tgt
    layernorm256<<<MQ, DD, 0, stream>>>(tmpB, n2_g, n2_b, t1);

    // ---- deformable cross attention ----
    addvec<<<MQ, DD, 0, stream>>>(t1, query_pos, qc, MQ * DD);
    gemm(qc, b_ca_so, ca_so_b, nullptr, offb, MQ, DD, DD, 0, MQ, 0, 0);
    gemm(qc, b_ca_aw, ca_aw_b, nullptr, awb, MQ, 128, DD, 0, MQ, 0, 0);
    softmax16<<<(BB * LQ * NH + 255) / 256, 256, 0, stream>>>(awb);
    gemm(src, b_ca_v, ca_v_b, nullptr, value, MV, DD, DD, 0, MV, 0, 0);     // reuse value buffer
    sample_cross<<<MQ * NH / 8, 256, 0, stream>>>(value, offb, awb, refpts, tmpA);
    gemm(tmpA, b_ca_out, ca_out_b, t1, tmpB, MQ, DD, DD, 0, MQ, 0, 0);      // + residual t1
    layernorm256<<<MQ, DD, 0, stream>>>(tmpB, n1_g, n1_b, qbuf);            // t2 (reuse qbuf)

    // ---- FFN ----
    gemm(qbuf, b_l1, l1_b, nullptr, ffn, MQ, DFF, DD, 1, MQ, 0, 0);         // relu
    gemm(ffn, b_l2, l2_b, qbuf, tmpB, MQ, DD, DFF, 0, MQ, 0, 0);            // + residual t2
    layernorm256<<<MQ, DD, 0, stream>>>(tmpB, n3_g, n3_b, (float*)d_out);
}
